// Model_10479720203003
// MI455X (gfx1250) — matrix operand order repaired and checked by reading, not compiled, not run on hardware
//
#include <hip/hip_runtime.h>
#include <hip/hip_bf16.h>

// ---------------------------------------------------------------------------
// MI455X (gfx1250, wave32) decoder layer.
// Every contraction runs through v_wmma_f32_16x16x32_bf16 (bf16 in, f32 acc).
// GEMM tiles are staged with GLOBAL_LOAD_ASYNC_TO_LDS_B128 (ASYNCcnt) in a
// double-buffered pipeline. Residual / softmax / layernorm math stays f32.
//
// Input order assumption (setup_inputs() insertion order, params flattened in
// insertion order):
//  0 char_encoder_output (64,128,1024) f32
//  1 word_encoder_output (64,32,1024)  f32
//  2 x                   (64,128,1024) f32
//  3 look_ahead_mask     (1,1,128,128) bool(u8)
//  4 char_padding_mask   (64,1,1,128)  bool(u8)
//  5 word_padding_mask   (64,1,1,32)   bool(u8)
//  6 sentence_lengths    (16,)         i32
//  7.. params: for blk in 1..3: wq,wk,wv,wo,pos,g,b ; then g4,b4,fw1,fb1,fw2,fb2
// ---------------------------------------------------------------------------

typedef unsigned short u16;
typedef __attribute__((ext_vector_type(16))) __bf16 v16bf;
typedef __attribute__((ext_vector_type(8)))  float  v8f;

union FragBF { v16bf v; unsigned int u[8]; };

__device__ __forceinline__ u16 f2bf(float f) {
  unsigned int u = __float_as_uint(f);
  u += 0x7FFFu + ((u >> 16) & 1u);        // round-to-nearest-even
  return (u16)(u >> 16);
}

__device__ __forceinline__ v8f wmma_bf16(const FragBF& a, const FragBF& b, v8f c) {
  return __builtin_amdgcn_wmma_f32_16x16x32_bf16(false, a.v, false, b.v,
                                                 (short)0, c, false, false);
}

// ---------------------------------------------------------------------------
// Generic batched bf16 GEMM:  C[z] = A[z] (MxK) * B[z] (NxK, pre-transposed)
// z decodes as (zb1 = z / zInner, zb2 = z % zInner); per-operand strides.
// Requirements: M % 128 == 0, K % 32 == 0, all operands 16B-aligned rows.
// N may be ragged (OOB B rows are read (harmless, inside ws) and their
// contributions discarded by the col<N store guard).
// trRows > 0: bf16 store scattered as [row/trRows][col/64][col%64][row%trRows]
// (builds the per-(batch,head) transposed V needed by the attn@V GEMM).
// ---------------------------------------------------------------------------
struct GemmP {
  const u16* A; long long sA1, sA2; int lda;
  const u16* B; long long sB1, sB2; int ldb;
  float* Cf; u16* Cb; long long sC1, sC2; int ldc;
  const float* bias;    // len N or null
  const float* resid;   // f32, same layout as C, or null
  int M, N, K, relu, zInner, trRows;
};

#define BM 128
#define BN 64
#define BKK 32

__global__ __launch_bounds__(256) void gemm_bf16_wmma(GemmP p) {
  __shared__ __align__(16) u16 As[2][BM * BKK];   // [m][k]
  __shared__ __align__(16) u16 Bs[2][BN * BKK];   // [n][k]

  const int z   = blockIdx.z;
  const int zb1 = z / p.zInner, zb2 = z % p.zInner;
  const u16* Ab = p.A + (long long)zb1 * p.sA1 + (long long)zb2 * p.sA2;
  const u16* Bb = p.B + (long long)zb1 * p.sB1 + (long long)zb2 * p.sB2;
  const long long coff = (long long)zb1 * p.sC1 + (long long)zb2 * p.sC2;

  const int m0 = blockIdx.y * BM;
  const int n0 = blockIdx.x * BN;
  const int t    = threadIdx.x;
  const int wave = t >> 5, lane = t & 31;
  const int wm = wave & 3;        // 4 row groups of 32
  const int wn = wave >> 2;       // 2 col groups of 32

  // per-thread 16B staging chunk: row lr (+64 for A's second half), cols lc..lc+7
  const int lr = t >> 2;          // 0..63
  const int lc = (t & 3) * 8;
  const unsigned aby0 = ((unsigned)(m0 + lr) * (unsigned)p.lda + (unsigned)lc) * 2u;
  const unsigned aby1 = aby0 + (unsigned)(64 * p.lda) * 2u;
  const unsigned bby  = ((unsigned)(n0 + lr) * (unsigned)p.ldb + (unsigned)lc) * 2u;
  const unsigned ldsA = (unsigned)(unsigned long long)&As[0][0];
  const unsigned ldsB = (unsigned)(unsigned long long)&Bs[0][0];
  const unsigned chunk = (unsigned)(lr * BKK + lc) * 2u;

  auto issue = [&](int ki, int buf) {
    const unsigned kb  = (unsigned)ki * (BKK * 2u);
    const unsigned la0 = ldsA + (unsigned)buf * (BM * BKK * 2u) + chunk;
    const unsigned la1 = la0 + 64u * BKK * 2u;
    const unsigned lb  = ldsB + (unsigned)buf * (BN * BKK * 2u) + chunk;
    const unsigned ga0 = aby0 + kb, ga1 = aby1 + kb, gb = bby + kb;
    asm volatile("global_load_async_to_lds_b128 %0, %1, %2"
                 :: "v"(la0), "v"(ga0), "s"(Ab) : "memory");
    asm volatile("global_load_async_to_lds_b128 %0, %1, %2"
                 :: "v"(la1), "v"(ga1), "s"(Ab) : "memory");
    asm volatile("global_load_async_to_lds_b128 %0, %1, %2"
                 :: "v"(lb),  "v"(gb),  "s"(Bb) : "memory");
  };

  v8f acc[2][2] = {};
  const int nk = p.K / BKK;

  issue(0, 0);
  for (int i = 0; i < nk; ++i) {
    const int buf = i & 1;
    if (i + 1 < nk) {
      issue(i + 1, buf ^ 1);
      // async loads retire in order: <=3 outstanding means tile i has landed
      asm volatile("s_wait_asynccnt 0x3" ::: "memory");
    } else {
      asm volatile("s_wait_asynccnt 0x0" ::: "memory");
    }
    __syncthreads();

    const u16* Ash = As[buf];
    const u16* Bsh = Bs[buf];
    FragBF fa0, fa1, fb0, fb1v;
    {
      // A 16x32 layout: lane<16 -> M=lane, K halves {0..7,16..23}; lane>=16 -> K+8
      const int r  = lane & 15;
      const int ka = (lane < 16) ? 0 : 8;
      const unsigned int* p00 = (const unsigned int*)&Ash[(wm * 32 + r) * BKK + ka];
      const unsigned int* p01 = (const unsigned int*)&Ash[(wm * 32 + r) * BKK + ka + 16];
      const unsigned int* p10 = (const unsigned int*)&Ash[(wm * 32 + 16 + r) * BKK + ka];
      const unsigned int* p11 = (const unsigned int*)&Ash[(wm * 32 + 16 + r) * BKK + ka + 16];
#pragma unroll
      for (int j = 0; j < 4; ++j) {
        fa0.u[j] = p00[j]; fa0.u[4 + j] = p01[j];
        fa1.u[j] = p10[j]; fa1.u[4 + j] = p11[j];
      }
    }
    {
      // B 32x16 layout: lane<16 -> N=lane, K halves {0..7,16..23}; lane>=16 -> K+8
      const int n  = lane & 15;
      const int kb2 = (lane < 16) ? 0 : 8;
      const unsigned int* q0 = (const unsigned int*)&Bsh[(wn * 32 + n) * BKK + kb2];
      const unsigned int* q1 = (const unsigned int*)&Bsh[(wn * 32 + 16 + n) * BKK + kb2];
#pragma unroll
      for (int j = 0; j < 8; ++j) { fb0.u[j] = q0[j < 4 ? j : j + 4]; fb1v.u[j] = q1[j < 4 ? j : j + 4]; }
    }

    acc[0][0] = wmma_bf16(fa0, fb0,  acc[0][0]);
    acc[0][1] = wmma_bf16(fa0, fb1v, acc[0][1]);
    acc[1][0] = wmma_bf16(fa1, fb0,  acc[1][0]);
    acc[1][1] = wmma_bf16(fa1, fb1v, acc[1][1]);
    __syncthreads();
  }

  // ---- epilogue: C layout: VGPR j -> M=j (lanes 0-15) / M=8+j (lanes 16-31)
  const int lcol  = lane & 15;
  const int lrow8 = (lane < 16) ? 0 : 8;
#pragma unroll
  for (int ri = 0; ri < 2; ++ri) {
#pragma unroll
    for (int ci = 0; ci < 2; ++ci) {
      const int col = n0 + wn * 32 + ci * 16 + lcol;
      if (col >= p.N) continue;
#pragma unroll
      for (int j = 0; j < 8; ++j) {
        const int row = m0 + wm * 32 + ri * 16 + lrow8 + j;
        if (row >= p.M) continue;
        float v = acc[ri][ci][j];
        if (p.bias)  v += p.bias[col];
        if (p.relu)  v = v > 0.f ? v : 0.f;
        const long long idx = coff + (long long)row * p.ldc + col;
        if (p.resid) v += p.resid[idx];
        if (p.Cf) p.Cf[idx] = v;
        if (p.Cb) {
          long long cbi = idx;
          if (p.trRows > 0) {
            const int rb = row / p.trRows, rk = row - rb * p.trRows;
            cbi = (((long long)rb * 16 + (col >> 6)) * 64 + (col & 63)) * p.trRows + rk;
          }
          p.Cb[cbi] = f2bf(v);
        }
      }
    }
  }
}

// ---------------------------------------------------------------------------
// f32 -> bf16 conversion
// ---------------------------------------------------------------------------
__global__ void cvt_f32_bf16(const float* __restrict__ in, u16* __restrict__ out,
                             long long n) {
  long long i = (long long)blockIdx.x * blockDim.x + threadIdx.x;
  const long long stride = (long long)gridDim.x * blockDim.x;
  for (; i < n; i += stride) out[i] = f2bf(in[i]);
}

// ---------------------------------------------------------------------------
// Tiled transpose + convert: in (K,N) f32 -> out (N,K) bf16.  K,N % 32 == 0.
// ---------------------------------------------------------------------------
__global__ __launch_bounds__(256) void transpose_cvt(const float* __restrict__ in,
                                                     u16* __restrict__ out,
                                                     int K, int N) {
  __shared__ float tile[32][33];
  const int kb = blockIdx.y * 32, nb = blockIdx.x * 32;
  const int tx = threadIdx.x & 31, ty = threadIdx.x >> 5;   // 32 x 8
#pragma unroll
  for (int i = 0; i < 32; i += 8)
    tile[ty + i][tx] = in[(long long)(kb + ty + i) * N + (nb + tx)];
  __syncthreads();
#pragma unroll
  for (int i = 0; i < 32; i += 8)
    out[(long long)(nb + ty + i) * K + (kb + tx)] = f2bf(tile[tx][ty + i]);
}

// ---------------------------------------------------------------------------
// Fused pos-gather + mask + scale + softmax.  One block per (b,h,q) row,
// blockDim.x == TK.  mode: 0=self(look-ahead), 1=cross-char, 2=cross-word.
// ---------------------------------------------------------------------------
__global__ void attn_softmax(const float* __restrict__ S,
                             const float* __restrict__ Pq,
                             const unsigned char* __restrict__ mask,
                             const int* __restrict__ slens,
                             u16* __restrict__ probs,
                             int mode, int TK) {
  __shared__ float red[128];
  const int row = blockIdx.x;
  const int k = threadIdx.x;
  const int q = row & 127;
  const int h = (row >> 7) & 15;
  const int b = row >> 11;
  const long long soff = (long long)row * TK;

  int anchor = q;
  if (mode == 2) {               // word position of batch element b
    int cum = 0, start = 0;
    for (int i = 0; i < 16; ++i) {
      const int len = slens[i];
      cum += len;
      if (b < start + len) { anchor = len - cum + b; break; }
      start += len;
    }
  }
  int idx = k - anchor;
  idx = idx < -64 ? -64 : (idx > 64 ? 64 : idx);
  idx += 64;
  const float pos = Pq[((long long)h * 8192 + (long long)b * 128 + q) * 129 + idx];

  bool msk;
  if (mode == 0) msk = mask[q * 128 + k] != 0;
  else           msk = mask[b * TK + k] != 0;

  float l = (S[soff + k] + pos) * 0.125f;   // 1/SCALE, SCALE = sqrt(64) = 8
  if (msk) l = -3.0e38f;

  red[k] = l; __syncthreads();
  for (int s = blockDim.x >> 1; s > 0; s >>= 1) {
    if (k < s) red[k] = fmaxf(red[k], red[k + s]);
    __syncthreads();
  }
  const float m = red[0]; __syncthreads();
  const float e = __expf(l - m);
  red[k] = e; __syncthreads();
  for (int s = blockDim.x >> 1; s > 0; s >>= 1) {
    if (k < s) red[k] += red[k + s];
    __syncthreads();
  }
  probs[soff + k] = f2bf(e / red[0]);
}

// ---------------------------------------------------------------------------
// LayerNorm over last dim (1024); writes f32 and optionally bf16.
// ---------------------------------------------------------------------------
__global__ __launch_bounds__(256) void layernorm_k(const float* __restrict__ X,
                                                   const float* __restrict__ gam,
                                                   const float* __restrict__ bet,
                                                   float* __restrict__ Yf,
                                                   u16* __restrict__ Yb) {
  __shared__ float s1[256], s2[256];
  const long long row = blockIdx.x;
  const float* xr = X + row * 1024;
  const int t = threadIdx.x;
  float a0 = xr[t], a1 = xr[t + 256], a2 = xr[t + 512], a3 = xr[t + 768];
  s1[t] = a0 + a1 + a2 + a3;
  s2[t] = a0 * a0 + a1 * a1 + a2 * a2 + a3 * a3;
  __syncthreads();
  for (int s = 128; s > 0; s >>= 1) {
    if (t < s) { s1[t] += s1[t + s]; s2[t] += s2[t + s]; }
    __syncthreads();
  }
  const float mu  = s1[0] * (1.f / 1024.f);
  const float var = s2[0] * (1.f / 1024.f) - mu * mu;
  const float rs  = rsqrtf(var + 1e-5f);
#pragma unroll
  for (int i = 0; i < 4; ++i) {
    const int c = t + i * 256;
    const float v = (xr[c] - mu) * rs * gam[c] + bet[c];
    if (Yf) Yf[row * 1024 + c] = v;
    if (Yb) Yb[row * 1024 + c] = f2bf(v);
  }
}

// ---------------------------------------------------------------------------
// Host-side helpers
// ---------------------------------------------------------------------------
static inline void gemm(hipStream_t st,
    const u16* A, long long sA1, long long sA2, int lda,
    const u16* B, long long sB1, long long sB2, int ldb,
    float* Cf, u16* Cb, long long sC1, long long sC2, int ldc,
    const float* bias, const float* resid,
    int M, int N, int K, int relu, int Z, int zInner, int trRows) {
  GemmP p;
  p.A = A; p.sA1 = sA1; p.sA2 = sA2; p.lda = lda;
  p.B = B; p.sB1 = sB1; p.sB2 = sB2; p.ldb = ldb;
  p.Cf = Cf; p.Cb = Cb; p.sC1 = sC1; p.sC2 = sC2; p.ldc = ldc;
  p.bias = bias; p.resid = resid;
  p.M = M; p.N = N; p.K = K; p.relu = relu; p.zInner = zInner; p.trRows = trRows;
  dim3 grid((N + BN - 1) / BN, (M + BM - 1) / BM, Z);
  gemm_bf16_wmma<<<grid, 256, 0, st>>>(p);
}

extern "C" void kernel_launch(void* const* d_in, const int* in_sizes, int n_in,
                              void* d_out, int out_size, void* d_ws, size_t ws_size,
                              hipStream_t stream) {
  (void)in_sizes; (void)n_in; (void)out_size; (void)ws_size;

  const int Dm = 1024, Hh = 16, DH = 64, TQ = 128, Bb = 64, TKC = 128, TKW = 32;
  const long long NQ  = (long long)Bb * TQ;        // 8192 query rows
  const long long NX  = NQ * Dm;
  const long long NCE = (long long)Bb * TKC * Dm;
  const long long NWE = (long long)Bb * TKW * Dm;

  const float* ceo  = (const float*)d_in[0];
  const float* weo  = (const float*)d_in[1];
  const float* x_in = (const float*)d_in[2];
  const unsigned char* la_mask = (const unsigned char*)d_in[3];
  const unsigned char* ch_mask = (const unsigned char*)d_in[4];
  const unsigned char* wd_mask = (const unsigned char*)d_in[5];
  const int* slens = (const int*)d_in[6];
  auto P = [&](int i) { return (const float*)d_in[7 + i]; };
  const float* g4  = P(21); const float* b4  = P(22);
  const float* fw1 = P(23); const float* fb1 = P(24);
  const float* fw2 = P(25); const float* fb2 = P(26);

  // ---- workspace bump allocator ----
  char* wsb = (char*)d_ws;
  size_t off = 0;
  auto alloc = [&](size_t bytes) -> void* {
    void* p = wsb + off;
    off = (off + bytes + 255) & ~(size_t)255;
    return p;
  };

  u16 *wqb[3], *wkb[3], *wvb[3], *wob[3], *posb[3];
  for (int i = 0; i < 3; ++i) {
    wqb[i]  = (u16*)alloc((size_t)Dm * Dm * 2);   // stored (N,K) = transposed
    wkb[i]  = (u16*)alloc((size_t)Dm * Dm * 2);
    wvb[i]  = (u16*)alloc((size_t)Dm * Dm * 2);
    wob[i]  = (u16*)alloc((size_t)Dm * Dm * 2);
    posb[i] = (u16*)alloc((size_t)129 * DH * 2);  // already (N,K)
  }
  u16* fw1b = (u16*)alloc((size_t)Dm * 4 * Dm * 2);  // (4096,1024)
  u16* fw2b = (u16*)alloc((size_t)Dm * 4 * Dm * 2);  // (1024,4096)
  u16* ceob = (u16*)alloc((size_t)NCE * 2);
  u16* weob = (u16*)alloc((size_t)NWE * 2);
  u16* x0b  = (u16*)alloc((size_t)NX * 2);
  float* xf_pp[2]  = {(float*)alloc((size_t)NX * 4), (float*)alloc((size_t)NX * 4)};
  u16*   xb_pp[2]  = {(u16*)alloc((size_t)NX * 2),   (u16*)alloc((size_t)NX * 2)};
  u16* Qb = (u16*)alloc((size_t)NX * 2);
  u16* Kb = (u16*)alloc((size_t)NX * 2);
  u16* Vt = (u16*)alloc((size_t)NX * 2);   // V stored [b][h][n=64][k=TK]
  float* Pq   = (float*)alloc((size_t)16 * 8192 * 129 * 4);
  float* Sraw = (float*)alloc((size_t)Bb * Hh * TQ * TKC * 4);
  u16*   probs = (u16*)alloc((size_t)Bb * Hh * TQ * TKC * 2);
  float* AOf = (float*)alloc((size_t)NX * 4);
  u16*   AOb = (u16*)alloc((size_t)NX * 2);
  float* tmp = (float*)alloc((size_t)NX * 4);
  u16* H1b = (u16*)Pq;   // alias: Pq (67.6MB) >= H1 (67.1MB); Pq dead by FFN time

  auto cvt = [&](const float* s, u16* d, long long n) {
    long long blk = (n + 255) / 256; if (blk > 4096) blk = 4096;
    cvt_f32_bf16<<<dim3((unsigned)blk), 256, 0, stream>>>(s, d, n);
  };
  auto tcvt = [&](const float* s, u16* d, int K, int N) {
    transpose_cvt<<<dim3(N / 32, K / 32), 256, 0, stream>>>(s, d, K, N);
  };

  // ---- one-time weight transposes / conversions to bf16 ----
  for (int i = 0; i < 3; ++i) {
    tcvt(P(i * 7 + 0), wqb[i], Dm, Dm);
    tcvt(P(i * 7 + 1), wkb[i], Dm, Dm);
    tcvt(P(i * 7 + 2), wvb[i], Dm, Dm);
    tcvt(P(i * 7 + 3), wob[i], Dm, Dm);
    cvt(P(i * 7 + 4), posb[i], (long long)129 * DH);
  }
  tcvt(fw1, fw1b, Dm, 4 * Dm);
  tcvt(fw2, fw2b, 4 * Dm, Dm);
  cvt(ceo, ceob, NCE);
  cvt(weo, weob, NWE);
  cvt(x_in, x0b, NX);

  const float* x_cur_f = x_in;
  const u16*   x_cur_b = x0b;
  const unsigned char* masks[3] = {la_mask, ch_mask, wd_mask};

  for (int blk = 0; blk < 3; ++blk) {
    const u16* kvb = (blk == 0) ? x_cur_b : (blk == 1 ? ceob : weob);
    const int  TK  = (blk == 2) ? TKW : TKC;
    const int  MKV = Bb * TK;

    // Q/K/V projections (weights pre-transposed -> (N,K))
    gemm(stream, x_cur_b, 0, 0, Dm, wqb[blk], 0, 0, Dm,
         nullptr, Qb, 0, 0, Dm, nullptr, nullptr, (int)NQ, Dm, Dm, 0, 1, 1, 0);
    gemm(stream, kvb, 0, 0, Dm, wkb[blk], 0, 0, Dm,
         nullptr, Kb, 0, 0, Dm, nullptr, nullptr, MKV, Dm, Dm, 0, 1, 1, 0);
    gemm(stream, kvb, 0, 0, Dm, wvb[blk], 0, 0, Dm,
         nullptr, Vt, 0, 0, Dm, nullptr, nullptr, MKV, Dm, Dm, 0, 1, 1, TK);

    // Pq[h, b*TQ+q, p] = Q_head . pos_emb[p]   (batched over h)
    gemm(stream, Qb, 0, 64, Dm, posb[blk], 0, 0, DH,
         Pq, nullptr, 0, (long long)NQ * 129, 129,
         nullptr, nullptr, (int)NQ, 129, DH, 0, Hh, Hh, 0);

    // S = Q K^T   (batched over (b,h); K rows are the (N,K) operand)
    gemm(stream, Qb, (long long)TQ * Dm, 64, Dm,
         Kb, (long long)TK * Dm, 64, Dm,
         Sraw, nullptr, (long long)Hh * TQ * TK, (long long)TQ * TK, TK,
         nullptr, nullptr, TQ, TK, DH, 0, Bb * Hh, Hh, 0);

    // fused pos-gather + mask + scale + softmax -> bf16 probs
    attn_softmax<<<dim3((unsigned)(Bb * Hh * TQ)), dim3((unsigned)TK), 0, stream>>>(
        Sraw, Pq, masks[blk], slens, probs, blk, TK);

    // O = probs @ V  (V already stored transposed per (b,h))
    gemm(stream, probs, (long long)Hh * TQ * TK, (long long)TQ * TK, TK,
         Vt, (long long)Hh * 64 * TK, (long long)64 * TK, TK,
         AOf, nullptr, (long long)TQ * Dm, 64, Dm,
         nullptr, nullptr, TQ, DH, TK, 0, Bb * Hh, Hh, 0);

    // output projection + residual, then layernorm -> next x (f32 + bf16)
    cvt(AOf, AOb, NX);
    gemm(stream, AOb, 0, 0, Dm, wob[blk], 0, 0, Dm,
         tmp, nullptr, 0, 0, Dm, nullptr, x_cur_f, (int)NQ, Dm, Dm, 0, 1, 1, 0);
    layernorm_k<<<dim3((unsigned)NQ), 256, 0, stream>>>(
        tmp, P(blk * 7 + 5), P(blk * 7 + 6), xf_pp[blk & 1], xb_pp[blk & 1]);
    x_cur_f = xf_pp[blk & 1];
    x_cur_b = xb_pp[blk & 1];
  }

  // ---- FFN: relu(x@fw1+fb1)@fw2+fb2, residual, final layernorm -> d_out ----
  gemm(stream, x_cur_b, 0, 0, Dm, fw1b, 0, 0, Dm,
       nullptr, H1b, 0, 0, 4 * Dm, fb1, nullptr, (int)NQ, 4 * Dm, Dm, 1, 1, 1, 0);
  gemm(stream, H1b, 0, 0, 4 * Dm, fw2b, 0, 0, 4 * Dm,
       tmp, nullptr, 0, 0, Dm, fb2, x_cur_f, (int)NQ, Dm, 4 * Dm, 0, 1, 1, 0);
  layernorm_k<<<dim3((unsigned)NQ), 256, 0, stream>>>(
      tmp, g4, b4, (float*)d_out, nullptr);
}
